// UserAgg_31267361915506
// MI455X (gfx1250) — matrix operand order repaired and checked by reading, not compiled, not run on hardware
//
#include <hip/hip_runtime.h>

typedef __attribute__((ext_vector_type(16))) _Float16 v16h;
typedef __attribute__((ext_vector_type(8)))  _Float16 v8h;
typedef __attribute__((ext_vector_type(8)))  float    v8f;

#define STAG_STRIDE 72   // halves per staged row: 64 + 8 pad => 144B, 16B-aligned

// ---- monotonic float<->uint encoding for atomicMax on signed floats ----
__device__ __forceinline__ unsigned enc_f32(float f) {
  unsigned u = __float_as_uint(f);
  return (u & 0x80000000u) ? ~u : (u | 0x80000000u);
}
__device__ __forceinline__ float dec_f32(unsigned v) {
  unsigned u = (v & 0x80000000u) ? (v & 0x7FFFFFFFu) : ~v;
  return __uint_as_float(u);
}

// ---- WMMA helper: D = A(16x32 f16) * B(32x16 f16) + C(f32) ----
#define WMMA(a, b, c) __builtin_amdgcn_wmma_f32_16x16x32_f16( \
    false, (a), false, (b), (short)0, (c), false, false)

// B-fragment stored fragment-major in LDS: 32 lanes x 16 halves (1KB per frag)
__device__ __forceinline__ v16h bfrag_from_lds(const _Float16* frag, int lane) {
  const v8h* p = (const v8h*)(frag + lane * 16);
  v8h lo = p[0], hi = p[1];
  v16h r;
#pragma unroll
  for (int i = 0; i < 8; ++i) { r[i] = lo[i]; r[8 + i] = hi[i]; }
  return r;
}

__device__ __forceinline__ void pack8(v16h& r, int base, float4 a, float4 b) {
  r[base+0]=(_Float16)a.x; r[base+1]=(_Float16)a.y; r[base+2]=(_Float16)a.z; r[base+3]=(_Float16)a.w;
  r[base+4]=(_Float16)b.x; r[base+5]=(_Float16)b.y; r[base+6]=(_Float16)b.z; r[base+7]=(_Float16)b.w;
}

// A-fragment (16x32, f16) built from a global f32 feature row. Per ISA layout:
// lane holds row m=lane&15; kb=(lane>=16)?8:0; v0..3 = K kb..kb+7, v4..7 = K 16+kb..16+kb+7
__device__ __forceinline__ v16h afrag_from_f32(const float* rowk, int kb) {
  float4 a0 = *(const float4*)(rowk + kb);
  float4 a1 = *(const float4*)(rowk + kb + 4);
  float4 b0 = *(const float4*)(rowk + 16 + kb);
  float4 b1 = *(const float4*)(rowk + 16 + kb + 4);
  v16h r; pack8(r, 0, a0, a1); pack8(r, 8, b0, b1); return r;
}

// A-fragment from f16 staging in LDS (row-major [16][STAG_STRIDE])
__device__ __forceinline__ v16h afrag_from_lds(const _Float16* stag, int m, int k0, int kb) {
  const _Float16* p = stag + m * STAG_STRIDE + k0 + kb;
  v8h lo = *(const v8h*)p;
  v8h hi = *(const v8h*)(p + 16);
  v16h r;
#pragma unroll
  for (int i = 0; i < 8; ++i) { r[i] = lo[i]; r[8 + i] = hi[i]; }
  return r;
}

// Write a C fragment (16x16 f32) as f16 into staging; lane holds N=lane&15, rows mb..mb+7
__device__ __forceinline__ void store_c_to_stag(_Float16* stag, v8f c, int lane, int nt, bool relu) {
  int n  = lane & 15;
  int mb = (lane >> 4) * 8;
#pragma unroll
  for (int r = 0; r < 8; ++r) {
    float v = c[r];
    if (relu) v = fmaxf(v, 0.0f);
    stag[(mb + r) * STAG_STRIDE + nt * 16 + n] = (_Float16)v;
  }
}

// Cooperative: convert a [K][64] f32 weight into fragment-major f16 LDS layout.
// frag = (k>>5)*4 + (n>>4); within frag: lane = (((k&31)>>3)&1)<<4 | (n&15); j = (k&7) | (((k&31)>>4)<<3)
__device__ __forceinline__ void build_wfrags(_Float16* dst, const float* W, int K, int tid, int nthr) {
  int total = K * 64;
  for (int idx = tid; idx < total; idx += nthr) {
    int k = idx >> 6;
    int ncol = idx & 63;
    int t = k >> 5;
    int kin = k & 31;
    int half = (kin >> 3) & 1;
    int j = (kin & 7) | ((kin >> 4) << 3);
    int lane = (half << 4) | (ncol & 15);
    int nt = ncol >> 4;
    int frag = t * 4 + nt;
    dst[frag * 512 + lane * 16 + j] = (_Float16)W[idx];
  }
}

// ---------------- kernel 0: init scratch ----------------
__global__ void init_ws_kernel(unsigned* __restrict__ m_enc, float* __restrict__ denom,
                               float* __restrict__ z, int I) {
  long i = (long)blockIdx.x * blockDim.x + threadIdx.x;
  long NZ = (long)I * 64;
  if (i < NZ) z[i] = 0.0f;
  if (i < I) { m_enc[i] = 0u; denom[i] = 0.0f; }
}

// ---------------- kernel 1: per-edge MLPs + attention score + segment max ----------------
__global__ void __launch_bounds__(256)
edge_fwd_kernel(const float* __restrict__ user_feat, const float* __restrict__ item_feat,
                const float* __restrict__ rating_feat,
                const int* __restrict__ row_idxs, const int* __restrict__ col_idxs,
                const int* __restrict__ rating,
                const float* __restrict__ gu_w1, const float* __restrict__ gu_b1,
                const float* __restrict__ gu_w2, const float* __restrict__ gu_b2,
                const float* __restrict__ att_w1, const float* __restrict__ att_b1,
                const float* __restrict__ att_w2, const float* __restrict__ att_b2,
                const float* __restrict__ att_w3, const float* __restrict__ att_b3,
                float* __restrict__ score_out, unsigned* __restrict__ m_enc, int E)
{
  extern __shared__ _Float16 smem[];
  _Float16* W1f = smem;                 // gu_w1: 16 frags
  _Float16* W2f = W1f + 16 * 512;       // gu_w2: 8 frags
  _Float16* A1f = W2f + 8 * 512;        // att_w1: 16 frags
  _Float16* A2f = A1f + 16 * 512;       // att_w2: 8 frags
  _Float16* stagbase = A2f + 8 * 512;

  int tid = threadIdx.x, nthr = blockDim.x;
  build_wfrags(W1f, gu_w1, 128, tid, nthr);
  build_wfrags(W2f, gu_w2, 64, tid, nthr);
  build_wfrags(A1f, att_w1, 128, tid, nthr);
  build_wfrags(A2f, att_w2, 64, tid, nthr);
  __syncthreads();

  int wave = tid >> 5, lane = tid & 31;
  _Float16* stagH = stagbase + wave * 2 * (16 * STAG_STRIDE);
  _Float16* stagF = stagH + 16 * STAG_STRIDE;

  int tile = blockIdx.x * (blockDim.x >> 5) + wave;
  long e0 = (long)tile * 16;
  if (e0 >= E) return;                    // wave-uniform

  int m = lane & 15;
  int n = lane & 15;
  int kb = (lane >> 4) * 8;
  int em = (int)e0 + m; if (em > E - 1) em = E - 1;
  int ur = row_idxs[em], rr = rating[em], ic = col_idxs[em];
  const float* urow = user_feat   + (size_t)ur * 64;
  const float* rrow = rating_feat + (size_t)rr * 64;
  const float* irow = item_feat   + (size_t)ic * 64;

  // x = [u | r], 16x128 as 4 A-frags
  v16h ax0 = afrag_from_f32(urow,      kb);
  v16h ax1 = afrag_from_f32(urow + 32, kb);
  v16h ax2 = afrag_from_f32(rrow,      kb);
  v16h ax3 = afrag_from_f32(rrow + 32, kb);

  // layer1: h = relu(x @ gu_w1 + b1) -> stagH
#pragma unroll
  for (int nt = 0; nt < 4; ++nt) {
    float bias = gu_b1[nt * 16 + n];
    v8f c;
#pragma unroll
    for (int r = 0; r < 8; ++r) c[r] = bias;
    c = WMMA(ax0, bfrag_from_lds(W1f + (0 * 4 + nt) * 512, lane), c);
    c = WMMA(ax1, bfrag_from_lds(W1f + (1 * 4 + nt) * 512, lane), c);
    c = WMMA(ax2, bfrag_from_lds(W1f + (2 * 4 + nt) * 512, lane), c);
    c = WMMA(ax3, bfrag_from_lds(W1f + (3 * 4 + nt) * 512, lane), c);
    store_c_to_stag(stagH, c, lane, nt, true);
  }

  // layer2: f = relu(h @ gu_w2 + b2) -> stagF
  v16h ah0 = afrag_from_lds(stagH, m, 0,  kb);
  v16h ah1 = afrag_from_lds(stagH, m, 32, kb);
#pragma unroll
  for (int nt = 0; nt < 4; ++nt) {
    float bias = gu_b2[nt * 16 + n];
    v8f c;
#pragma unroll
    for (int r = 0; r < 8; ++r) c[r] = bias;
    c = WMMA(ah0, bfrag_from_lds(W2f + nt * 512, lane), c);
    c = WMMA(ah1, bfrag_from_lds(W2f + (4 + nt) * 512, lane), c);
    store_c_to_stag(stagF, c, lane, nt, true);
  }

  // attention input [f | i_emb]
  v16h af0 = afrag_from_lds(stagF, m, 0,  kb);
  v16h af1 = afrag_from_lds(stagF, m, 32, kb);
  v16h ai0 = afrag_from_f32(irow,      kb);
  v16h ai1 = afrag_from_f32(irow + 32, kb);

  // att layer1 -> stagH (reuse)
#pragma unroll
  for (int nt = 0; nt < 4; ++nt) {
    float bias = att_b1[nt * 16 + n];
    v8f c;
#pragma unroll
    for (int r = 0; r < 8; ++r) c[r] = bias;
    c = WMMA(af0, bfrag_from_lds(A1f + nt * 512, lane), c);
    c = WMMA(af1, bfrag_from_lds(A1f + (4 + nt) * 512, lane), c);
    c = WMMA(ai0, bfrag_from_lds(A1f + (8 + nt) * 512, lane), c);
    c = WMMA(ai1, bfrag_from_lds(A1f + (12 + nt) * 512, lane), c);
    store_c_to_stag(stagH, c, lane, nt, true);
  }

  // att layer2 + fused dot with att_w3 (score per row)
  v16h aa0 = afrag_from_lds(stagH, m, 0,  kb);
  v16h aa1 = afrag_from_lds(stagH, m, 32, kb);
  float part[8];
#pragma unroll
  for (int r = 0; r < 8; ++r) part[r] = 0.0f;
#pragma unroll
  for (int nt = 0; nt < 4; ++nt) {
    float bias = att_b2[nt * 16 + n];
    v8f c;
#pragma unroll
    for (int r = 0; r < 8; ++r) c[r] = bias;
    c = WMMA(aa0, bfrag_from_lds(A2f + nt * 512, lane), c);
    c = WMMA(aa1, bfrag_from_lds(A2f + (4 + nt) * 512, lane), c);
    float w3 = att_w3[nt * 16 + n];
#pragma unroll
    for (int r = 0; r < 8; ++r) part[r] += fmaxf(c[r], 0.0f) * w3;
  }
  // reduce across the 16 lanes of each half (wave32 shuffles)
#pragma unroll
  for (int off = 8; off >= 1; off >>= 1) {
#pragma unroll
    for (int r = 0; r < 8; ++r) part[r] += __shfl_xor(part[r], off, 32);
  }

  float b3 = att_b3[0];
  if ((lane & 15) == 0) {
    int mb = (lane >> 4) * 8;
#pragma unroll
    for (int r = 0; r < 8; ++r) {
      long e = e0 + mb + r;
      if (e < E) {
        float s = part[r] + b3;
        score_out[e] = s;
        atomicMax(&m_enc[col_idxs[e]], enc_f32(s));
      }
    }
  }
}

// ---------------- kernel 2: e = exp(score - m[col]); denom += e ----------------
__global__ void softmax_denom_kernel(const int* __restrict__ col_idxs,
                                     const unsigned* __restrict__ m_enc,
                                     float* __restrict__ escore,
                                     float* __restrict__ denom, int E) {
  long e = (long)blockIdx.x * blockDim.x + threadIdx.x;
  if (e >= E) return;
  int c = col_idxs[e];
  float mv = dec_f32(m_enc[c]);
  float ev = __expf(escore[e] - mv);
  escore[e] = ev;
  atomicAdd(&denom[c], ev);
}

// ---------------- kernel 3: recompute f_jt, scatter z[col] += f * alpha ----------------
__global__ void __launch_bounds__(256)
scatter_kernel(const float* __restrict__ user_feat, const float* __restrict__ rating_feat,
               const int* __restrict__ row_idxs, const int* __restrict__ col_idxs,
               const int* __restrict__ rating,
               const float* __restrict__ gu_w1, const float* __restrict__ gu_b1,
               const float* __restrict__ gu_w2, const float* __restrict__ gu_b2,
               const float* __restrict__ escore, const float* __restrict__ denom,
               float* __restrict__ z, int E)
{
  extern __shared__ _Float16 smem[];
  _Float16* W1f = smem;
  _Float16* W2f = W1f + 16 * 512;
  _Float16* stagbase = W2f + 8 * 512;

  int tid = threadIdx.x, nthr = blockDim.x;
  build_wfrags(W1f, gu_w1, 128, tid, nthr);
  build_wfrags(W2f, gu_w2, 64, tid, nthr);
  __syncthreads();

  int wave = tid >> 5, lane = tid & 31;
  _Float16* stagH = stagbase + wave * 2 * (16 * STAG_STRIDE);
  _Float16* stagF = stagH + 16 * STAG_STRIDE;

  int tile = blockIdx.x * (blockDim.x >> 5) + wave;
  long e0 = (long)tile * 16;
  if (e0 >= E) return;

  int m = lane & 15;
  int n = lane & 15;
  int kb = (lane >> 4) * 8;
  int em = (int)e0 + m; if (em > E - 1) em = E - 1;
  int ur = row_idxs[em], rr = rating[em];
  const float* urow = user_feat   + (size_t)ur * 64;
  const float* rrow = rating_feat + (size_t)rr * 64;

  v16h ax0 = afrag_from_f32(urow,      kb);
  v16h ax1 = afrag_from_f32(urow + 32, kb);
  v16h ax2 = afrag_from_f32(rrow,      kb);
  v16h ax3 = afrag_from_f32(rrow + 32, kb);

#pragma unroll
  for (int nt = 0; nt < 4; ++nt) {
    float bias = gu_b1[nt * 16 + n];
    v8f c;
#pragma unroll
    for (int r = 0; r < 8; ++r) c[r] = bias;
    c = WMMA(ax0, bfrag_from_lds(W1f + (0 * 4 + nt) * 512, lane), c);
    c = WMMA(ax1, bfrag_from_lds(W1f + (1 * 4 + nt) * 512, lane), c);
    c = WMMA(ax2, bfrag_from_lds(W1f + (2 * 4 + nt) * 512, lane), c);
    c = WMMA(ax3, bfrag_from_lds(W1f + (3 * 4 + nt) * 512, lane), c);
    store_c_to_stag(stagH, c, lane, nt, true);
  }

  v16h ah0 = afrag_from_lds(stagH, m, 0,  kb);
  v16h ah1 = afrag_from_lds(stagH, m, 32, kb);
#pragma unroll
  for (int nt = 0; nt < 4; ++nt) {
    float bias = gu_b2[nt * 16 + n];
    v8f c;
#pragma unroll
    for (int r = 0; r < 8; ++r) c[r] = bias;
    c = WMMA(ah0, bfrag_from_lds(W2f + nt * 512, lane), c);
    c = WMMA(ah1, bfrag_from_lds(W2f + (4 + nt) * 512, lane), c);
    store_c_to_stag(stagF, c, lane, nt, true);
  }

  // per-row alpha & destination col into (now free) stagH
  int*   colA = (int*)stagH;
  float* alpA = (float*)(colA + 16);
  if (lane < 16) {
    long e = e0 + lane;
    int c = 0; float a = 0.0f;
    if (e < E) { c = col_idxs[e]; a = escore[e] / denom[c]; }
    colA[lane] = c; alpA[lane] = a;
  }

  int d0 = lane * 2;
  for (int mm = 0; mm < 16; ++mm) {
    if (e0 + mm >= E) break;            // uniform
    float a = alpA[mm];
    int   c = colA[mm];
    float f0 = (float)stagF[mm * STAG_STRIDE + d0];
    float f1 = (float)stagF[mm * STAG_STRIDE + d0 + 1];
    atomicAdd(&z[(size_t)c * 64 + d0],     f0 * a);
    atomicAdd(&z[(size_t)c * 64 + d0 + 1], f1 * a);
  }
}

// ---------------- kernel 4: out = z @ w_w + w_b ----------------
__global__ void __launch_bounds__(256)
outproj_kernel(const float* __restrict__ z, const float* __restrict__ w_w,
               const float* __restrict__ w_b, float* __restrict__ out, int I)
{
  int lane = threadIdx.x & 31, wave = threadIdx.x >> 5;
  int nn = lane & 15;
  int khalf = (lane >> 4) * 16;
  int kb = (lane >> 4) * 8;
  int m = lane & 15;
  int mb = (lane >> 4) * 8;

  // w_w B-fragments in registers: 2 ktiles x 4 ntiles = 8 frags = 64 VGPRs
  v16h bw[8];
#pragma unroll
  for (int t = 0; t < 2; ++t)
#pragma unroll
    for (int nt = 0; nt < 4; ++nt) {
      v16h b;
#pragma unroll
      for (int j = 0; j < 16; ++j)
        b[j] = (_Float16)w_w[(size_t)(t * 32 + kb + (j & 7) + ((j >> 3) << 4)) * 64 + nt * 16 + nn];
      bw[t * 4 + nt] = b;
    }

  int tilesTotal = (I + 15) / 16;
  int nW = gridDim.x * (blockDim.x >> 5);
  for (int tile = blockIdx.x * (blockDim.x >> 5) + wave; tile < tilesTotal; tile += nW) {
    int row0 = tile * 16;
    int rm = row0 + m; if (rm > I - 1) rm = I - 1;
    const float* zrow = z + (size_t)rm * 64;
    v16h az0 = afrag_from_f32(zrow,      kb);
    v16h az1 = afrag_from_f32(zrow + 32, kb);
#pragma unroll
    for (int nt = 0; nt < 4; ++nt) {
      float bias = w_b[nt * 16 + nn];
      v8f c;
#pragma unroll
      for (int r = 0; r < 8; ++r) c[r] = bias;
      c = WMMA(az0, bw[nt], c);
      c = WMMA(az1, bw[4 + nt], c);
#pragma unroll
      for (int r = 0; r < 8; ++r) {
        int row = row0 + mb + r;
        if (row < I) out[(size_t)row * 64 + nt * 16 + nn] = c[r];
      }
    }
  }
}

extern "C" void kernel_launch(void* const* d_in, const int* in_sizes, int n_in,
                              void* d_out, int out_size, void* d_ws, size_t ws_size,
                              hipStream_t stream)
{
  const float* user_feat   = (const float*)d_in[0];
  const float* item_feat   = (const float*)d_in[1];
  const float* rating_feat = (const float*)d_in[2];
  const int*   row_idxs    = (const int*)d_in[3];
  const int*   col_idxs    = (const int*)d_in[4];
  const int*   rating      = (const int*)d_in[5];
  const float* gu_w1  = (const float*)d_in[6];
  const float* gu_b1  = (const float*)d_in[7];
  const float* gu_w2  = (const float*)d_in[8];
  const float* gu_b2  = (const float*)d_in[9];
  const float* att_w1 = (const float*)d_in[10];
  const float* att_b1 = (const float*)d_in[11];
  const float* att_w2 = (const float*)d_in[12];
  const float* att_b2 = (const float*)d_in[13];
  const float* att_w3 = (const float*)d_in[14];
  const float* att_b3 = (const float*)d_in[15];
  const float* w_w    = (const float*)d_in[16];
  const float* w_b    = (const float*)d_in[17];

  int E = in_sizes[3];
  int I = in_sizes[1] / 64;

  // workspace: [escore: E f32][m_enc: I u32][denom: I f32][z: I*64 f32] ~= 17 MB
  float*    escore = (float*)d_ws;
  unsigned* m_enc  = (unsigned*)(escore + E);
  float*    denom  = (float*)(m_enc + I);
  float*    z      = denom + I;
  float*    out    = (float*)d_out;

  {
    long N = (long)I * 64;
    int gr = (int)((N + 255) / 256);
    init_ws_kernel<<<gr, 256, 0, stream>>>(m_enc, denom, z, I);
  }

  int tiles = (E + 15) / 16;
  int blocksT = (tiles + 7) / 8;   // 8 waves per 256-thread block
  size_t smemA = (size_t)(48 * 512 + 8 * 2 * 16 * STAG_STRIDE) * sizeof(_Float16);
  edge_fwd_kernel<<<blocksT, 256, smemA, stream>>>(
      user_feat, item_feat, rating_feat, row_idxs, col_idxs, rating,
      gu_w1, gu_b1, gu_w2, gu_b2, att_w1, att_b1, att_w2, att_b2,
      att_w3, att_b3, escore, m_enc, E);

  {
    int gr = (E + 255) / 256;
    softmax_denom_kernel<<<gr, 256, 0, stream>>>(col_idxs, m_enc, escore, denom, E);
  }

  size_t smemC = (size_t)(24 * 512 + 8 * 2 * 16 * STAG_STRIDE) * sizeof(_Float16);
  scatter_kernel<<<blocksT, 256, smemC, stream>>>(
      user_feat, rating_feat, row_idxs, col_idxs, rating,
      gu_w1, gu_b1, gu_w2, gu_b2, escore, denom, z, E);

  {
    int tilesI = (I + 15) / 16;
    int blocksD = (tilesI + 7) / 8;
    outproj_kernel<<<blocksD, 256, 0, stream>>>(z, w_w, w_b, out, I);
  }
}
